// DC_ECC_Transformer_16183436772162
// MI455X (gfx1250) — matrix operand order repaired and checked by reading, not compiled, not run on hardware
//
#include <hip/hip_runtime.h>
#include <hip/hip_bf16.h>

// ---------------- problem constants ----------------
#define BB    4
#define NC    1024
#define NKD   512
#define DD    128
#define HH    8
#define NLAYER 6
#define DKH   16
#define LL    1536          // NC + NKD
#define DFFD  512
#define DHH   50
#define LUTN  1024

typedef __attribute__((ext_vector_type(16))) _Float16 v16h;
typedef __attribute__((ext_vector_type(8)))  _Float16 v8h;
typedef __attribute__((ext_vector_type(4)))  _Float16 v4h;
typedef __attribute__((ext_vector_type(8)))  float    v8f;
typedef __attribute__((ext_vector_type(4)))  float    v4f;

union F16x16 { v16h v; v8h h[2]; };
__device__ __forceinline__ v16h mk16(v8h lo, v8h hi){ F16x16 u; u.h[0] = lo; u.h[1] = hi; return u.v; }

__device__ __forceinline__ float sgnf(float v){ return (v > 0.f) ? 1.f : ((v < 0.f) ? -1.f : 0.f); }
__device__ __forceinline__ float gelu_tanh(float x){
  float x3 = x * x * x;
  return 0.5f * x * (1.f + tanhf(0.7978845608028654f * (x + 0.044715f * x3)));
}

// ---------------- workspace layout (bytes) ----------------
static constexpr size_t SZ_X    = (size_t)BB * LL * DD * 4;
static constexpr size_t SZ_M    = (size_t)LL * LL * 4;
static constexpr size_t SZ_MASK = (size_t)LL * LL * 4;
static constexpr size_t SZ_H16  = (size_t)BB * LL * DD * 2;
static constexpr size_t SZ_HH   = (size_t)BB * LL * DFFD * 2;
static constexpr size_t SZ_W    = (size_t)NLAYER * DD * DD * 2;
static constexpr size_t SZ_FW   = (size_t)NLAYER * DD * DFFD * 2;
static constexpr size_t SZ_AGG  = (size_t)BB * NC * DD * 4;
static constexpr size_t SZ_LG   = (size_t)BB * NC * 4;
static constexpr size_t SZ_PCH  = (size_t)NKD * NC * 2;

static constexpr size_t OFF_X    = 0;
static constexpr size_t OFF_M    = OFF_X    + SZ_X;
static constexpr size_t OFF_MASK = OFF_M    + SZ_M;
static constexpr size_t OFF_XN   = OFF_MASK + SZ_MASK;
static constexpr size_t OFF_Q    = OFF_XN   + SZ_H16;
static constexpr size_t OFF_K    = OFF_Q    + SZ_H16;
static constexpr size_t OFF_V    = OFF_K    + SZ_H16;   // stored transposed: vT[(b,h,dk)][l]
static constexpr size_t OFF_O    = OFF_V    + SZ_H16;
static constexpr size_t OFF_HH   = OFF_O    + SZ_H16;
static constexpr size_t OFF_WQT  = OFF_HH   + SZ_HH;    // transposed f16 weights
static constexpr size_t OFF_WKT  = OFF_WQT  + SZ_W;
static constexpr size_t OFF_WVT  = OFF_WKT  + SZ_W;
static constexpr size_t OFF_WOT  = OFF_WVT  + SZ_W;
static constexpr size_t OFF_FW1T = OFF_WOT  + SZ_W;
static constexpr size_t OFF_FW2T = OFF_FW1T + SZ_FW;
static constexpr size_t OFF_AGG  = OFF_FW2T + SZ_FW;
static constexpr size_t OFF_ZP   = OFF_AGG  + SZ_AGG;
static constexpr size_t OFF_LG   = OFF_ZP   + SZ_AGG;
static constexpr size_t OFF_PCH  = OFF_LG   + ((SZ_LG + 255) & ~(size_t)255);
static constexpr size_t OFF_PCT  = OFF_PCH  + SZ_PCH;
static constexpr size_t OFF_LUT  = OFF_PCT  + SZ_PCH;                       // 4 KB
static constexpr size_t OFF_ZMH  = OFF_LUT  + LUTN * 4;                     // f16 (B*NC, D)
static constexpr size_t OFF_ZST  = OFF_ZMH  + (size_t)BB * NC * DD * 2;     // f16 (B*D, NK)
static constexpr size_t OFF_AGGH = OFF_ZST  + (size_t)BB * DD * NKD * 2;    // f16 (B*NC, D)
static constexpr size_t OFF_B2   = OFF_AGGH + (size_t)BB * NC * DD * 2;     // 128 f32
static constexpr size_t OFF_MWT  = OFF_B2   + 512;
static constexpr size_t OFF_SWT  = OFF_MWT  + (size_t)DD * DD * 2;

// ---------------- conversion kernels ----------------
__global__ void f32_to_f16_kernel(const float* __restrict__ a, _Float16* __restrict__ o, int n){
  int i = blockIdx.x * blockDim.x + threadIdx.x;
  if (i < n) o[i] = (_Float16)a[i];
}

// per-layer blocks: src (K x N) row-major -> dst[n*K + k]  (transposed f16)
__global__ void conv_transpose_kernel(const float* __restrict__ src, _Float16* __restrict__ dst,
                                      int K, int N, int total){
  int idx = blockIdx.x * blockDim.x + threadIdx.x;
  if (idx >= total) return;
  int per = K * N;
  int l = idx / per, rem = idx % per;
  int k = rem / N, n = rem % N;
  dst[(size_t)l * per + (size_t)n * K + k] = (_Float16)src[idx];
}

__global__ void embed_kernel(const float* __restrict__ mag, const int* __restrict__ synd,
                             const float* __restrict__ magn_emb, const float* __restrict__ synd_tab,
                             float* __restrict__ x){
  int idx = blockIdx.x * blockDim.x + threadIdx.x;
  if (idx >= BB * LL * DD) return;
  int d = idx % DD;
  int l = (idx / DD) % LL;
  int b = idx / (DD * LL);
  float v;
  if (l < NC) v = magn_emb[d] * mag[b * NC + l];
  else        v = synd_tab[synd[b * NKD + (l - NC)] * DD + d];
  x[idx] = v;
}

// off-diagonal blocks of M: top-right = P^T, bottom-left = P
__global__ void m_offdiag_kernel(const float* __restrict__ pc, float* __restrict__ Mb){
  int idx = blockIdx.x * blockDim.x + threadIdx.x;
  const int tot = NKD * NC;
  if (idx >= 2 * tot) return;
  int which = idx / tot, rem = idx % tot;
  int kr = rem / NC, c = rem % NC;
  float v = pc[kr * NC + c];
  if (which == 0) Mb[(size_t)c * LL + NC + kr]   = v;
  else            Mb[(size_t)(NC + kr) * LL + c] = v;
}

// M is integer-valued -> evaluate mask MLP once per value (scale 1/sqrt(DK) folded in)
__global__ __launch_bounds__(256)
void mask_lut_kernel(const float* __restrict__ w1, const float* __restrict__ b1,
                     const float* __restrict__ w2, const float* __restrict__ b2,
                     float* __restrict__ lut){
  int v = blockIdx.x * 256 + threadIdx.x;
  if (v >= LUTN) return;
  float m = (float)v;
  float acc = b2[0];
  #pragma unroll 10
  for (int h = 0; h < DHH; ++h)
    acc += fmaxf(m * w1[h] + b1[h], 0.f) * w2[h];
  lut[v] = 0.25f * acc;
}

__global__ __launch_bounds__(256)
void mask_apply_kernel(const float* __restrict__ Mb, const float* __restrict__ lut,
                       float* __restrict__ maskb){
  __shared__ float luts[LUTN];
  int t = threadIdx.x;
  #pragma unroll
  for (int i = 0; i < LUTN / 256; ++i) luts[t + i * 256] = lut[t + i * 256];
  __syncthreads();
  size_t idx = ((size_t)blockIdx.x * 256 + t) * 4;
  v4f mv = *(const v4f*)(Mb + idx);
  v4f o;
  #pragma unroll
  for (int j = 0; j < 4; ++j) {
    int vi = (int)(mv[j] + 0.5f);
    vi = vi < 0 ? 0 : (vi > LUTN - 1 ? LUTN - 1 : vi);
    o[j] = luts[vi];
  }
  *(v4f*)(maskb + idx) = o;
}

// one wave per row; shuffle reductions.  MODE 0: f16 out; MODE 1: f32 out
template<int MODE>
__global__ __launch_bounds__(32)
void ln_kernel(const float* __restrict__ xin, const float* __restrict__ g,
               const float* __restrict__ be, _Float16* __restrict__ oh, float* __restrict__ of){
  const int row = blockIdx.x, lane = threadIdx.x;
  v4f xv = *(const v4f*)(xin + (size_t)row * DD + lane * 4);
  float s = xv[0] + xv[1] + xv[2] + xv[3];
  s += __shfl_xor(s, 16, 32); s += __shfl_xor(s, 8, 32); s += __shfl_xor(s, 4, 32);
  s += __shfl_xor(s, 2, 32);  s += __shfl_xor(s, 1, 32);
  const float mean = s * (1.f / DD);
  v4f dv; float vs = 0.f;
  #pragma unroll
  for (int j = 0; j < 4; ++j) { dv[j] = xv[j] - mean; vs += dv[j] * dv[j]; }
  vs += __shfl_xor(vs, 16, 32); vs += __shfl_xor(vs, 8, 32); vs += __shfl_xor(vs, 4, 32);
  vs += __shfl_xor(vs, 2, 32);  vs += __shfl_xor(vs, 1, 32);
  const float rstd = rsqrtf(vs * (1.f / DD) + 1e-5f);
  v4f gv = *(const v4f*)(g + lane * 4);
  v4f bv = *(const v4f*)(be + lane * 4);
  if (MODE == 0) {
    v4h o;
    #pragma unroll
    for (int j = 0; j < 4; ++j) o[j] = (_Float16)(dv[j] * rstd * gv[j] + bv[j]);
    *(v4h*)(oh + (size_t)row * DD + lane * 4) = o;
  } else {
    v4f o;
    #pragma unroll
    for (int j = 0; j < 4; ++j) o[j] = dv[j] * rstd * gv[j] + bv[j];
    *(v4f*)(of + (size_t)row * DD + lane * 4) = o;
  }
}

// ---------------- WMMA GEMM (two 16-row tiles per wave; B fragment reused) ----------------
// MODE 0: f16 + bias | 1: f16 + bias + gelu | 2: f32 += with bias | 3: vT scatter f16 + bias
// MODE 4: f32 store no bias | 5: f32 store + bias | 6: f32 += no bias
template<int MODE>
__global__ __launch_bounds__(32)
void gemm_kernel(const _Float16* __restrict__ A, const _Float16* __restrict__ BT,
                 const float* __restrict__ bias, _Float16* __restrict__ Ch,
                 float* __restrict__ Cf, int M, int N, int K, int ldc){
  const int lane = threadIdx.x;
  const int tn = blockIdx.x, tm2 = blockIdx.y;
  const int n   = lane & 15;
  const int col = tn * 16 + n;
  const int kbA = (lane >> 4) * 8;
  const int kbB = (lane >> 4) * 16;
  const _Float16* arow0 = A + (size_t)(tm2 * 32 + n) * K;
  const _Float16* arow1 = arow0 + (size_t)16 * K;
  const _Float16* brow  = BT + (size_t)col * K;
  v8f c0 = {}, c1 = {};
  for (int k0 = 0; k0 < K; k0 += 32) {
    v8h b0 = *(const v8h*)(brow + k0 + kbA);
    v8h b1 = *(const v8h*)(brow + k0 + kbA + 16);
    v16h bf = mk16(b0, b1);
    v8h a00 = *(const v8h*)(arow0 + k0 + kbA);
    v8h a01 = *(const v8h*)(arow0 + k0 + kbA + 16);
    c0 = __builtin_amdgcn_wmma_f32_16x16x32_f16(false, mk16(a00, a01), false, bf, (short)0, c0, false, false);
    v8h a10 = *(const v8h*)(arow1 + k0 + kbA);
    v8h a11 = *(const v8h*)(arow1 + k0 + kbA + 16);
    c1 = __builtin_amdgcn_wmma_f32_16x16x32_f16(false, mk16(a10, a11), false, bf, (short)0, c1, false, false);
  }
  const float bv = (MODE == 4 || MODE == 6) ? 0.f : bias[col];
  const int rhalf = (lane >> 4) * 8;
  if (MODE == 3) {
    const int h = col >> 4, dk = col & 15;
    const int gr0 = tm2 * 32;
    const int bidx = gr0 / LL, l0 = (gr0 % LL) + rhalf;
    const size_t base = ((size_t)(bidx * HH + h) * DKH + dk) * LL;
    v8h out0, out1;
    #pragma unroll
    for (int r = 0; r < 8; ++r) { out0[r] = (_Float16)(c0[r] + bv); out1[r] = (_Float16)(c1[r] + bv); }
    *(v8h*)(Ch + base + l0)      = out0;
    *(v8h*)(Ch + base + l0 + 16) = out1;
  } else {
    #pragma unroll
    for (int t = 0; t < 2; ++t) {
      #pragma unroll
      for (int r = 0; r < 8; ++r) {
        int row = tm2 * 32 + t * 16 + r + rhalf;
        float v = (t == 0 ? c0[r] : c1[r]) + bv;
        if (MODE == 1) v = gelu_tanh(v);
        if (MODE == 2 || MODE == 6) Cf[(size_t)row * ldc + col] += v;
        else if (MODE == 4 || MODE == 5) Cf[(size_t)row * ldc + col] = v;
        else Ch[(size_t)row * ldc + col] = (_Float16)v;
      }
    }
  }
}

// ---------------- flash attention: 32 keys per iteration ----------------
// 2 QK^T WMMAs (16 real keys each, K-padded) + 1 full-K PV WMMA (32 real keys).
// Softmax via LDS transpose; lane (row n, half) owns keys [kbA,kbA+8) u [kbA+16,kbA+24)
// == exactly the WMMA A-fragment layout, so the P fragment is built in registers.
__global__ __launch_bounds__(32)
void attention_kernel(const _Float16* __restrict__ qh, const _Float16* __restrict__ kh,
                      const _Float16* __restrict__ vT, const float* __restrict__ maskb,
                      _Float16* __restrict__ oh){
  const int lane = threadIdx.x;
  const int qt = blockIdx.x;
  const int h  = blockIdx.y;
  const int b  = blockIdx.z;

  __shared__ __align__(16) float sbuf[16 * 32];
  __shared__ __align__(16) float abuf[16];

  const int n     = lane & 15;
  const int rhalf = (lane >> 4) * 8;   // == kbA
  const int kbB   = (lane >> 4) * 16;
  const v8h zero8 = {};

  const size_t qbase = (size_t)(b * LL + qt * 16 + n) * DD + h * DKH;
  const v16h aq = (lane < 16) ? mk16(*(const v8h*)(qh + qbase), *(const v8h*)(qh + qbase + 8)) : mk16(zero8, zero8);

  float m_i = -1e30f, l_i = 0.f;
  v8f o = {};
  for (int kt = 0; kt < LL / 32; ++kt) {
    const int kbase = kt * 32;
    if (kt + 1 < LL / 32)
      __builtin_prefetch(kh + (size_t)(b * LL + kbase + 32 + n) * DD + h * DKH, 0, 1);

    // two K^T B-fragments (16 keys each, K-pad in upper half)
    v16h bk0, bk1;
    if (lane < 16) {
      const _Float16* kp0 = kh + (size_t)(b * LL + kbase + n) * DD + h * DKH;
      const _Float16* kp1 = kp0 + (size_t)16 * DD;
      bk0 = mk16(*(const v8h*)kp0, *(const v8h*)(kp0 + 8));
      bk1 = mk16(*(const v8h*)kp1, *(const v8h*)(kp1 + 8));
    } else { bk0 = mk16(zero8, zero8); bk1 = mk16(zero8, zero8); }

    v8f s0 = {}, s1 = {};
    s0 = __builtin_amdgcn_wmma_f32_16x16x32_f16(false, aq, false, bk0, (short)0, s0, false, false);
    s1 = __builtin_amdgcn_wmma_f32_16x16x32_f16(false, aq, false, bk1, (short)0, s1, false, false);

    // mask (symmetric, scale pre-folded): column reads, 8 contiguous floats per column
    const float* mp0 = maskb + (size_t)(kbase + n) * LL + qt * 16 + rhalf;
    const float* mp1 = mp0 + (size_t)16 * LL;
    v4f m00 = *(const v4f*)mp0, m01 = *(const v4f*)(mp0 + 4);
    v4f m10 = *(const v4f*)mp1, m11 = *(const v4f*)(mp1 + 4);
    #pragma unroll
    for (int r = 0; r < 4; ++r) {
      sbuf[(r + rhalf) * 32 + n]          = s0[r]     * m00[r];
      sbuf[(r + 4 + rhalf) * 32 + n]      = s0[r + 4] * m01[r];
      sbuf[(r + rhalf) * 32 + 16 + n]     = s1[r]     * m10[r];
      sbuf[(r + 4 + rhalf) * 32 + 16 + n] = s1[r + 4] * m11[r];
    }
    __syncthreads();

    // transposed view: row n, 16 owned keys in two contiguous chunks
    const float* srow = &sbuf[n * 32];
    v4f x0 = *(const v4f*)(srow + rhalf);
    v4f x1 = *(const v4f*)(srow + rhalf + 4);
    v4f x2 = *(const v4f*)(srow + 16 + rhalf);
    v4f x3 = *(const v4f*)(srow + 16 + rhalf + 4);
    float lmax = -1e30f;
    #pragma unroll
    for (int j = 0; j < 4; ++j)
      lmax = fmaxf(lmax, fmaxf(fmaxf(x0[j], x1[j]), fmaxf(x2[j], x3[j])));
    lmax = fmaxf(lmax, __shfl_xor(lmax, 16, 32));
    float mnew  = fmaxf(m_i, lmax);
    float alpha = __expf(m_i - mnew);
    float p[16]; float lsum = 0.f;
    #pragma unroll
    for (int j = 0; j < 4; ++j) {
      p[j]      = __expf(x0[j] - mnew); lsum += p[j];
      p[4 + j]  = __expf(x1[j] - mnew); lsum += p[4 + j];
      p[8 + j]  = __expf(x2[j] - mnew); lsum += p[8 + j];
      p[12 + j] = __expf(x3[j] - mnew); lsum += p[12 + j];
    }
    lsum += __shfl_xor(lsum, 16, 32);
    l_i = l_i * alpha + lsum;
    m_i = mnew;
    if (lane < 16) abuf[lane] = alpha;

    // full P A-fragment (no padding): element j<8 -> key kbA+j, j>=8 -> key kbA+8+j
    v16h pa;
    #pragma unroll
    for (int j = 0; j < 8; ++j) { pa[j] = (_Float16)p[j]; pa[8 + j] = (_Float16)p[8 + j]; }
    __syncthreads();

    v4f a0 = *(const v4f*)&abuf[rhalf];
    v4f a1 = *(const v4f*)&abuf[rhalf + 4];
    #pragma unroll
    for (int r = 0; r < 4; ++r) { o[r] *= a0[r]; o[r + 4] *= a1[r]; }

    // V B-fragment: 32 real keys, keys [kbA,kbA+8) u [kbA+16,kbA+24) per lane
    const _Float16* vp = vT + ((size_t)(b * HH + h) * DKH + n) * LL + kbase + rhalf;
    v16h bvf = mk16(*(const v8h*)vp, *(const v8h*)(vp + 16));

    o = __builtin_amdgcn_wmma_f32_16x16x32_f16(false, pa, false, bvf, (short)0, o, false, false);
    __syncthreads();
  }

  if (lane < 16) abuf[lane] = l_i;
  __syncthreads();
  v4f l0 = *(const v4f*)&abuf[rhalf];
  v4f l1 = *(const v4f*)&abuf[rhalf + 4];
  #pragma unroll
  for (int r = 0; r < 8; ++r) {
    int row = r + rhalf;
    float val = o[r] / (r < 4 ? l0[r] : l1[r - 4]);
    oh[(size_t)(b * LL + qt * 16 + row) * DD + h * DKH + n] = (_Float16)val;
  }
}

// ---------------- head ----------------
__global__ void zm_convert_kernel(const float* __restrict__ emb, _Float16* __restrict__ zmH){
  int idx = blockIdx.x * blockDim.x + threadIdx.x;
  if (idx >= BB * NC * DD) return;
  int d = idx % DD, nn = (idx / DD) % NC, b = idx / (NC * DD);
  zmH[idx] = (_Float16)(-emb[((size_t)b * LL + nn) * DD + d]);
}

__global__ void zsT_kernel(const float* __restrict__ emb, _Float16* __restrict__ zsT){
  int idx = blockIdx.x * blockDim.x + threadIdx.x;
  if (idx >= BB * DD * NKD) return;
  int c = idx % NKD, d = (idx / NKD) % DD, b = idx / (DD * NKD);
  zsT[idx] = (_Float16)(-emb[((size_t)b * LL + NC + c) * DD + d]);
}

__global__ void bias2_kernel(const float* __restrict__ mag_b, const float* __restrict__ synd_b,
                             float* __restrict__ b2){
  int d = threadIdx.x;
  if (d < DD) b2[d] = mag_b[d] + (float)NKD * synd_b[d];
}

__global__ void logits_kernel(const float* __restrict__ zp, const float* __restrict__ out_w,
                              const float* __restrict__ out_b, float* __restrict__ lg){
  int idx = blockIdx.x * blockDim.x + threadIdx.x;
  if (idx >= BB * NC) return;
  float acc = out_b[0];
  const float* row = zp + (size_t)idx * DD;
  for (int d = 0; d < DD; ++d) acc += row[d] * out_w[d];
  lg[idx] = acc;
}

__global__ __launch_bounds__(256)
void final_kernel(const float* __restrict__ lg, const float* __restrict__ z_mul,
                  const float* __restrict__ yv, float* __restrict__ loss, float* __restrict__ pred){
  __shared__ float red[256];
  int t = threadIdx.x;
  float s = 0.f;
  for (int i = t; i < BB * NC; i += 256) {
    float l = lg[i];
    float tt = 0.5f * (1.f - sgnf(z_mul[i]));
    s += fmaxf(l, 0.f) - l * tt + log1pf(__expf(-fabsf(l)));
    pred[i] = 0.5f * (1.f - sgnf(-l * sgnf(yv[i])));
  }
  red[t] = s; __syncthreads();
  for (int k = 128; k > 0; k >>= 1) { if (t < k) red[t] += red[t + k]; __syncthreads(); }
  if (t == 0) loss[0] = red[0] / (float)(BB * NC);
}

// ---------------- launcher ----------------
extern "C" void kernel_launch(void* const* d_in, const int* in_sizes, int n_in,
                              void* d_out, int out_size, void* d_ws, size_t ws_size,
                              hipStream_t stream) {
  (void)in_sizes; (void)n_in; (void)out_size; (void)ws_size;

  const float* magnitude = (const float*)d_in[0];
  const int*   syndrome  = (const int*)d_in[1];
  const float* pc        = (const float*)d_in[2];
  const float* z_mul     = (const float*)d_in[3];
  const float* yv        = (const float*)d_in[4];
  const float* magn_emb  = (const float*)d_in[5];
  const float* synd_tab  = (const float*)d_in[6];
  const float* Wq = (const float*)d_in[7],  *bq = (const float*)d_in[8];
  const float* Wk = (const float*)d_in[9],  *bk = (const float*)d_in[10];
  const float* Wv = (const float*)d_in[11], *bv = (const float*)d_in[12];
  const float* Wo = (const float*)d_in[13], *bo = (const float*)d_in[14];
  const float* m_w1 = (const float*)d_in[15], *m_b1 = (const float*)d_in[16];
  const float* m_w2 = (const float*)d_in[17], *m_b2 = (const float*)d_in[18];
  const float* ln_a_g = (const float*)d_in[19], *ln_a_b = (const float*)d_in[20];
  const float* ln_b_g = (const float*)d_in[21], *ln_b_b = (const float*)d_in[22];
  const float* f_w1 = (const float*)d_in[23], *f_b1 = (const float*)d_in[24];
  const float* f_w2 = (const float*)d_in[25], *f_b2 = (const float*)d_in[26];
  const float* mid_g = (const float*)d_in[27], *mid_b = (const float*)d_in[28];
  const float* fin_g = (const float*)d_in[29], *fin_b = (const float*)d_in[30];
  const float* mag_w = (const float*)d_in[31], *mag_b = (const float*)d_in[32];
  const float* synd_w = (const float*)d_in[33], *synd_b = (const float*)d_in[34];
  const float* out_w = (const float*)d_in[35], *out_b = (const float*)d_in[36];

  char* w = (char*)d_ws;
  float*    x     = (float*)   (w + OFF_X);
  float*    Mb    = (float*)   (w + OFF_M);
  float*    maskb = (float*)   (w + OFF_MASK);
  _Float16* xn    = (_Float16*)(w + OFF_XN);
  _Float16* qh    = (_Float16*)(w + OFF_Q);
  _Float16* kh    = (_Float16*)(w + OFF_K);
  _Float16* vT    = (_Float16*)(w + OFF_V);
  _Float16* oh    = (_Float16*)(w + OFF_O);
  _Float16* hh    = (_Float16*)(w + OFF_HH);
  _Float16* wqT   = (_Float16*)(w + OFF_WQT);
  _Float16* wkT   = (_Float16*)(w + OFF_WKT);
  _Float16* wvT   = (_Float16*)(w + OFF_WVT);
  _Float16* woT   = (_Float16*)(w + OFF_WOT);
  _Float16* fw1T  = (_Float16*)(w + OFF_FW1T);
  _Float16* fw2T  = (_Float16*)(w + OFF_FW2T);
  float*    agg   = (float*)   (w + OFF_AGG);
  float*    zp    = (float*)   (w + OFF_ZP);
  float*    lg    = (float*)   (w + OFF_LG);
  _Float16* pcH   = (_Float16*)(w + OFF_PCH);
  _Float16* pcT   = (_Float16*)(w + OFF_PCT);
  float*    lut   = (float*)   (w + OFF_LUT);
  _Float16* zmH   = (_Float16*)(w + OFF_ZMH);
  _Float16* zsT   = (_Float16*)(w + OFF_ZST);
  _Float16* aggH  = (_Float16*)(w + OFF_AGGH);
  float*    b2    = (float*)   (w + OFF_B2);
  _Float16* mwT   = (_Float16*)(w + OFF_MWT);
  _Float16* swT   = (_Float16*)(w + OFF_SWT);

  float* emb  = (float*)d_out;
  float* loss = emb + (size_t)BB * LL * DD;
  float* pred = loss + 1;

  int nW  = NLAYER * DD * DD;
  int nFW = NLAYER * DD * DFFD;
  conv_transpose_kernel<<<(nW  + 255) / 256, 256, 0, stream>>>(Wq,   wqT,  DD,   DD,   nW);
  conv_transpose_kernel<<<(nW  + 255) / 256, 256, 0, stream>>>(Wk,   wkT,  DD,   DD,   nW);
  conv_transpose_kernel<<<(nW  + 255) / 256, 256, 0, stream>>>(Wv,   wvT,  DD,   DD,   nW);
  conv_transpose_kernel<<<(nW  + 255) / 256, 256, 0, stream>>>(Wo,   woT,  DD,   DD,   nW);
  conv_transpose_kernel<<<(nFW + 255) / 256, 256, 0, stream>>>(f_w1, fw1T, DD,   DFFD, nFW);
  conv_transpose_kernel<<<(nFW + 255) / 256, 256, 0, stream>>>(f_w2, fw2T, DFFD, DD,   nFW);
  conv_transpose_kernel<<<(DD * DD + 255) / 256, 256, 0, stream>>>(mag_w,  mwT, DD, DD, DD * DD);
  conv_transpose_kernel<<<(DD * DD + 255) / 256, 256, 0, stream>>>(synd_w, swT, DD, DD, DD * DD);
  f32_to_f16_kernel<<<(NKD * NC + 255) / 256, 256, 0, stream>>>(pc, pcH, NKD * NC);
  conv_transpose_kernel<<<(NKD * NC + 255) / 256, 256, 0, stream>>>(pc, pcT, NKD, NC, NKD * NC);

  embed_kernel<<<(BB * LL * DD + 255) / 256, 256, 0, stream>>>(magnitude, syndrome, magn_emb, synd_tab, x);

  const dim3 blk32(32);
  gemm_kernel<4><<<dim3(NC / 16, NC / 32), blk32, 0, stream>>>(pcT, pcT, nullptr, nullptr,
                                                               Mb, NC, NC, NKD, LL);
  gemm_kernel<4><<<dim3(NKD / 16, NKD / 32), blk32, 0, stream>>>(pcH, pcH, nullptr, nullptr,
                                                                 Mb + (size_t)NC * LL + NC, NKD, NKD, NC, LL);
  m_offdiag_kernel<<<(2 * NKD * NC + 255) / 256, 256, 0, stream>>>(pc, Mb);

  const int M = BB * LL;
  const dim3 gqkv(DD / 16, M / 32);
  const dim3 gff1(DFFD / 16, M / 32);

  for (int i = 0; i < NLAYER; ++i) {
    mask_lut_kernel<<<LUTN / 256, 256, 0, stream>>>(m_w1 + i * DHH, m_b1 + i * DHH,
                                                    m_w2 + i * DHH, m_b2 + i, lut);
    mask_apply_kernel<<<(LL * LL) / 1024, 256, 0, stream>>>(Mb, lut, maskb);
    ln_kernel<0><<<M, blk32, 0, stream>>>(x, ln_a_g + i * DD, ln_a_b + i * DD, xn, nullptr);

    gemm_kernel<0><<<gqkv, blk32, 0, stream>>>(xn, wqT + (size_t)i * DD * DD, bq + i * DD, qh, nullptr, M, DD, DD, DD);
    gemm_kernel<0><<<gqkv, blk32, 0, stream>>>(xn, wkT + (size_t)i * DD * DD, bk + i * DD, kh, nullptr, M, DD, DD, DD);
    gemm_kernel<3><<<gqkv, blk32, 0, stream>>>(xn, wvT + (size_t)i * DD * DD, bv + i * DD, vT, nullptr, M, DD, DD, DD);

    attention_kernel<<<dim3(LL / 16, HH, BB), blk32, 0, stream>>>(qh, kh, vT, maskb, oh);

    gemm_kernel<2><<<gqkv, blk32, 0, stream>>>(oh, woT + (size_t)i * DD * DD, bo + i * DD, nullptr, x, M, DD, DD, DD);

    ln_kernel<0><<<M, blk32, 0, stream>>>(x, ln_b_g + i * DD, ln_b_b + i * DD, xn, nullptr);
    gemm_kernel<1><<<gff1, blk32, 0, stream>>>(xn, fw1T + (size_t)i * DD * DFFD, f_b1 + i * DFFD, hh, nullptr, M, DFFD, DD, DFFD);
    gemm_kernel<2><<<gqkv, blk32, 0, stream>>>(hh, fw2T + (size_t)i * DFFD * DD, f_b2 + i * DD, nullptr, x, M, DD, DFFD, DD);

    if (i == 2) ln_kernel<1><<<M, blk32, 0, stream>>>(x, mid_g, mid_b, nullptr, x);
  }

  ln_kernel<1><<<M, blk32, 0, stream>>>(x, fin_g, fin_b, nullptr, emb);

  // head: agg_b = P^T @ zs_b ; zp = zm@mag_w + agg@synd_w + (mag_b + NK*synd_b)
  zm_convert_kernel<<<(BB * NC * DD + 255) / 256, 256, 0, stream>>>(emb, zmH);
  zsT_kernel<<<(BB * DD * NKD + 255) / 256, 256, 0, stream>>>(emb, zsT);
  for (int b = 0; b < BB; ++b)
    gemm_kernel<4><<<dim3(DD / 16, NC / 32), blk32, 0, stream>>>(pcT, zsT + (size_t)b * DD * NKD,
                                                                 nullptr, nullptr,
                                                                 agg + (size_t)b * NC * DD, NC, DD, NKD, DD);
  f32_to_f16_kernel<<<(BB * NC * DD + 255) / 256, 256, 0, stream>>>(agg, aggH, BB * NC * DD);
  bias2_kernel<<<1, 128, 0, stream>>>(mag_b, synd_b, b2);
  gemm_kernel<5><<<dim3(DD / 16, (BB * NC) / 32), blk32, 0, stream>>>(zmH, mwT, b2, nullptr, zp, BB * NC, DD, DD, DD);
  gemm_kernel<6><<<dim3(DD / 16, (BB * NC) / 32), blk32, 0, stream>>>(aggH, swT, nullptr, nullptr, zp, BB * NC, DD, DD, DD);
  logits_kernel<<<(BB * NC + 127) / 128, 128, 0, stream>>>(zp, out_w, out_b, lg);
  final_kernel<<<1, 256, 0, stream>>>(lg, z_mul, yv, loss, pred);
}
